// CroNDBase_68375879352472
// MI455X (gfx1250) — compile-verified, matrix operand order checked by reading, not run on hardware
//
#include <hip/hip_runtime.h>
#include <hip/hip_bf16.h>

// ---------------------------------------------------------------------------
// Model dims (from reference)
// ---------------------------------------------------------------------------
#define BS_    128
#define P_     16
#define M_     64
#define C_     21
#define H_     256
#define HO_    16
#define FLAT_  5376          // HO * P * C
#define ROWS_  2048          // BS * HO
#define KTILES_ 168          // FLAT / 32

typedef __attribute__((ext_vector_type(16))) __bf16 v16bf;
typedef __attribute__((ext_vector_type(8)))  __bf16 v8bf;
typedef __attribute__((ext_vector_type(8)))  float  v8f;

__device__ __forceinline__ unsigned short f2bf_u16(float f) {
  // round-to-nearest-even f32 -> bf16
  unsigned u = __float_as_uint(f);
  unsigned r = u + 0x7FFFu + ((u >> 16) & 1u);
  return (unsigned short)(r >> 16);
}

// ---------------------------------------------------------------------------
// Kernel 1: im2col + Gaussian RBF histogram + log/tanh, fused transpose,
// bf16 downconvert. One block per (batch, paper); 64x64 tile staged in LDS.
// Output layout x[b*16+ho][(c*16+p)*16+wo]  (row-major, FLAT_ cols, bf16)
// ---------------------------------------------------------------------------
__global__ void __launch_bounds__(256)
hist_kernel(const float* __restrict__ sim, unsigned short* __restrict__ xbf) {
  __shared__ float tile[M_ * M_];          // 16 KB
  const int bp = blockIdx.x;
  const int b = bp >> 4, p = bp & 15;
  const float* src = sim + (size_t)bp * (M_ * M_);
  const int t = threadIdx.x;

  for (int i = t; i < M_ * M_; i += 256) tile[i] = src[i];
  __syncthreads();

  // 5376 = 21 bins * 256 patches items per (b,p); 21 items per thread
  for (int item = t; item < C_ * HO_ * HO_; item += 256) {
    const int c   = item >> 8;
    const int rem = item & 255;
    const int ho  = rem >> 4, wo = rem & 15;
    // bin centers: mus[0]=1.0, mus[c>=1] = 0.95 - 0.1*(c-1)
    const float mu = (c == 0) ? 1.0f : (0.95f - 0.1f * (float)(c - 1));
    float rbf = 0.0f;
    const int base = (ho * 4) * M_ + wo * 4;
#pragma unroll
    for (int ki = 0; ki < 4; ++ki)
#pragma unroll
      for (int kj = 0; kj < 4; ++kj) {
        const float d = tile[base + ki * M_ + kj] - mu;
        rbf += __expf(-50.0f * d * d);   // 1/(2*0.1^2) = 50
      }
    const float g = tanhf(0.01f * __logf(fmaxf(rbf, 1e-10f)));
    const size_t idx = (size_t)(b * HO_ + ho) * FLAT_ +
                       (size_t)((c * P_ + p) * HO_ + wo);
    xbf[idx] = f2bf_u16(g);
  }
}

// ---------------------------------------------------------------------------
// Kernel 2: W_flat[5376,256] f32 -> W^T[256][5376] bf16 (B fragments load
// exactly like A fragments: two contiguous 16B chunks per lane)
// ---------------------------------------------------------------------------
__global__ void __launch_bounds__(256)
pack_w_kernel(const float* __restrict__ W, unsigned short* __restrict__ wt) {
  const int o = blockIdx.x * 256 + threadIdx.x;   // o = n*FLAT + k
  const int n = o / FLAT_;
  const int k = o - n * FLAT_;
  wt[o] = f2bf_u16(W[(size_t)k * H_ + n]);
}

// ---------------------------------------------------------------------------
// Kernel 3: h1 = tanh(x @ W_flat + b_flat) via v_wmma_f32_16x16x32_bf16.
// One wave32 per 16x16 output tile; 168 K-steps.
// A lane layout (16-bit A 16x32): lanes 0-15 -> K{0..7,16..23} of row lane,
// lanes 16-31 -> K{8..15,24..31} of row lane-16. B^T rows load identically.
// ---------------------------------------------------------------------------
__global__ void __launch_bounds__(32)
gemm_kernel(const unsigned short* __restrict__ xbf,
            const unsigned short* __restrict__ wt,
            const float* __restrict__ bflat,
            float* __restrict__ h1) {
  const int lane = threadIdx.x;
  const int mt = blockIdx.x;        // 0..127 row tiles
  const int nt = blockIdx.y;        // 0..15  col tiles
  const int half = lane >> 4;       // 0 or 1
  const int l15  = lane & 15;

  const unsigned short* arow = xbf + (size_t)(mt * 16 + l15) * FLAT_ + half * 8;
  const unsigned short* brow = wt  + (size_t)(nt * 16 + l15) * FLAT_ + half * 8;

  v8f acc = {};
  for (int kt = 0; kt < KTILES_; ++kt) {
    const v8bf* ap = (const v8bf*)(arow + kt * 32);
    const v8bf* bp = (const v8bf*)(brow + kt * 32);
    const v8bf a0 = ap[0], a1 = ap[2];   // +0 elems, +16 elems
    const v8bf b0 = bp[0], b1 = bp[2];
    v16bf A, B;
#pragma unroll
    for (int i = 0; i < 8; ++i) {
      A[i] = a0[i]; A[i + 8] = a1[i];
      B[i] = b0[i]; B[i + 8] = b1[i];
    }
    acc = __builtin_amdgcn_wmma_f32_16x16x32_bf16(
        /*neg_a=*/false, A, /*neg_b=*/false, B,
        /*c_mod=*/(short)0, acc, /*reuse_a=*/false, /*reuse_b=*/false);
  }

  // C/D layout: VGPR i -> row (i + 8*half), col l15
  const int col = nt * 16 + l15;
  const float bias = bflat[col];
#pragma unroll
  for (int i = 0; i < 8; ++i) {
    const int row = mt * 16 + half * 8 + i;
    h1[(size_t)row * H_ + col] = tanhf(acc[i] + bias);
  }
}

// ---------------------------------------------------------------------------
// Kernel 4: attention pooling + f_add projection + 2 dense layers +
// log_softmax. One block (256 threads) per batch element, all in LDS.
// d_out layout: [0..255] log_softmax (128x2), [256..511] raw logits (128x2)
// ---------------------------------------------------------------------------
__global__ void __launch_bounds__(256)
head_kernel(const float* __restrict__ h1,   const float* __restrict__ fadd,
            const float* __restrict__ Wattn, const float* __restrict__ battn,
            const float* __restrict__ Wfadd, const float* __restrict__ bfadd,
            const float* __restrict__ Wd1,   const float* __restrict__ bd1,
            const float* __restrict__ Wd2,   const float* __restrict__ bd2,
            const float* __restrict__ wfeat, float* __restrict__ out) {
  __shared__ float sh_h1[HO_ * H_];   // 16 KB
  __shared__ float sh_attn[HO_];
  __shared__ float sh_fa[17];
  __shared__ float sh_comb[H_];
  __shared__ float sh_score[H_];
  __shared__ float sh_o[2];

  const int b = blockIdx.x, t = threadIdx.x;

  for (int ho = 0; ho < HO_; ++ho)
    sh_h1[ho * H_ + t] = h1[(size_t)(b * HO_ + ho) * H_ + t];
  if (t < 17) sh_fa[t] = fadd[b * 17 + t];
  __syncthreads();

  // attention logits over the 16 positions
  if (t < HO_) {
    float s = battn[0];
    for (int j = 0; j < H_; ++j) s += sh_h1[t * H_ + j] * Wattn[j];
    sh_attn[t] = s;
  }
  __syncthreads();
  if (t == 0) {
    float mx = sh_attn[0];
    for (int i = 1; i < HO_; ++i) mx = fmaxf(mx, sh_attn[i]);
    float e[HO_], den = 0.0f;
    for (int i = 0; i < HO_; ++i) { e[i] = __expf(sh_attn[i] - mx); den += e[i]; }
    const float inv = 1.0f / den;
    for (int i = 0; i < HO_; ++i) sh_attn[i] = e[i] * inv;
  }
  __syncthreads();

  float lc = 0.0f;
  for (int ho = 0; ho < HO_; ++ho) lc += sh_attn[ho] * sh_h1[ho * H_ + t];
  lc = tanhf(lc);

  float fp = bfadd[t];
  for (int j = 0; j < 17; ++j) fp += sh_fa[j] * Wfadd[j * H_ + t];
  fp = tanhf(fp);

  const float s = 1.0f / (1.0f + __expf(-wfeat[0]));
  sh_comb[t] = (1.0f - s) * fp + s * lc;
  __syncthreads();

  float sc = bd1[t];
  for (int j = 0; j < H_; ++j) sc += sh_comb[j] * Wd1[j * H_ + t];
  sh_score[t] = tanhf(sc);
  __syncthreads();

  if (t < 2) {
    float o = bd2[t];
    for (int j = 0; j < H_; ++j) o += sh_score[j] * Wd2[j * 2 + t];
    sh_o[t] = o;
  }
  __syncthreads();
  if (t == 0) {
    const float o0 = sh_o[0], o1 = sh_o[1];
    const float mx = fmaxf(o0, o1);
    const float lse = mx + __logf(__expf(o0 - mx) + __expf(o1 - mx));
    out[b * 2 + 0] = o0 - lse;
    out[b * 2 + 1] = o1 - lse;
    out[BS_ * 2 + b * 2 + 0] = o0;
    out[BS_ * 2 + b * 2 + 1] = o1;
  }
}

// ---------------------------------------------------------------------------
extern "C" void kernel_launch(void* const* d_in, const int* in_sizes, int n_in,
                              void* d_out, int out_size, void* d_ws, size_t ws_size,
                              hipStream_t stream) {
  const float* sim   = (const float*)d_in[0];
  const float* fadd  = (const float*)d_in[1];
  const float* Wflat = (const float*)d_in[2];
  const float* bflat = (const float*)d_in[3];
  const float* Wattn = (const float*)d_in[4];
  const float* battn = (const float*)d_in[5];
  const float* Wfadd = (const float*)d_in[6];
  const float* bfadd = (const float*)d_in[7];
  const float* Wd1   = (const float*)d_in[8];
  const float* bd1   = (const float*)d_in[9];
  const float* Wd2   = (const float*)d_in[10];
  const float* bd2   = (const float*)d_in[11];
  const float* wf    = (const float*)d_in[12];
  float* out = (float*)d_out;

  // workspace carve-up (all 16B aligned)
  char* ws = (char*)d_ws;
  unsigned short* xbf = (unsigned short*)ws;                                   // 2048*5376 bf16
  unsigned short* wt  = (unsigned short*)(ws + (size_t)ROWS_ * FLAT_ * 2);     // 256*5376 bf16
  float* h1 = (float*)(ws + (size_t)ROWS_ * FLAT_ * 2 + (size_t)H_ * FLAT_ * 2); // 2048*256 f32

  hist_kernel  <<<dim3(BS_ * P_),            dim3(256), 0, stream>>>(sim, xbf);
  pack_w_kernel<<<dim3((H_ * FLAT_) / 256),  dim3(256), 0, stream>>>(Wflat, wt);
  gemm_kernel  <<<dim3(ROWS_ / 16, H_ / 16), dim3(32),  0, stream>>>(xbf, wt, bflat, h1);
  head_kernel  <<<dim3(BS_),                 dim3(256), 0, stream>>>(
      h1, fadd, Wattn, battn, Wfadd, bfadd, Wd1, bd1, Wd2, bd2, wf, out);
}
